// AnisotropicHomogeneousNN_4501125726631
// MI455X (gfx1250) — matrix operand order repaired and checked by reading, not compiled, not run on hardware
//
#include <hip/hip_runtime.h>

// Fused AnisotropicHomogeneousNN forward for MI455X (gfx1250, wave32).
//
// Roofline: GEMMs are 17.2 GFLOP over 134 MB of mandatory HBM traffic
// (x in, out out) -> compute-bound on matrix cores; everything between the
// input load and the output store (x_sphere, H=[B,256]) is kept in LDS so
// no intermediate touches HBM. Both GEMMs run on v_wmma_f32_16x16x32_f16.

typedef __attribute__((ext_vector_type(16))) _Float16 v16h;
typedef __attribute__((ext_vector_type(8)))  _Float16 v8h;
typedef __attribute__((ext_vector_type(8)))  float    v8f;

#define NEWTON_TOL 1e-7f
#define NEWTON_ITERS 30

constexpr int Nd   = 64;    // input dim
constexpr int Hd   = 256;   // hidden dim
constexpr int Od   = 64;    // output dim
constexpr int ROWS = 256;   // batch rows per workgroup (1 row / thread in phase 1)
constexpr int XS_S = 72;    // LDS stride (halves) for x_sphere & W1 tiles (64 + pad, 16B-aligned)
constexpr int H_S  = 264;   // LDS stride (halves) for H & W2 tiles (256 + pad, 16B-aligned)

constexpr size_t SMEM_BYTES =
    (size_t)(ROWS * XS_S + Hd * XS_S + ROWS * H_S + Od * H_S) * 2   // f16 tiles
  + (size_t)(ROWS + Hd + Od) * 4;                                    // scale + b1 + b2

// A fragment (16x32 f16, MxK) from row-major LDS tile.
// ISA 7.12.2: lanes 0-15 hold K={0..7,16..23}, lanes 16-31 K={8..15,24..31},
// row M = lane%16, packed 2 halves per VGPR in K order.
__device__ inline v16h ld_a_frag(const _Float16* base, int stride, int k0, int lane) {
  const int m  = lane & 15;
  const int h8 = (lane >> 4) << 3;
  const _Float16* p = base + m * stride + k0 + h8;
  v8h lo = *(const v8h*)(p);        // K = k0 + h8 + 0..7
  v8h hi = *(const v8h*)(p + 16);   // K = k0 + 16 + h8 + 0..7
  return __builtin_shufflevector(lo, hi, 0,1,2,3,4,5,6,7,8,9,10,11,12,13,14,15);
}

// B fragment (32x16 f16, KxN). Column n of B = row n of the weight tile
// (W stored as [out][in], so B[k][n] = W[n][k] is contiguous in k).
// Lanes 0-15 hold K = k0+{0..7,16..23} of column n0+lane, lanes 16-31 hold K = k0+{8..15,24..31}.
__device__ inline v16h ld_b_frag(const _Float16* w, int col0, int stride, int k0, int lane) {
  const _Float16* p = w + (col0 + (lane & 15)) * stride + k0 + ((lane >> 4) << 3);
  v8h lo = *(const v8h*)(p);
  v8h hi = *(const v8h*)(p + 16);
  return __builtin_shufflevector(lo, hi, 0,1,2,3,4,5,6,7,8,9,10,11,12,13,14,15);
}

extern "C" __global__ __launch_bounds__(256, 1)
void aniso_mlp_fused(const float* __restrict__ x,  const float* __restrict__ r,
                     const float* __restrict__ W1, const float* __restrict__ b1,
                     const float* __restrict__ W2, const float* __restrict__ b2,
                     float* __restrict__ out) {
  extern __shared__ char smem_raw[];
  _Float16* sX  = (_Float16*)smem_raw;          // [ROWS][XS_S] x_sphere (f16)
  _Float16* sW1 = sX  + ROWS * XS_S;            // [Hd][XS_S]   W1 rows  (f16)
  _Float16* sH  = sW1 + Hd   * XS_S;            // [ROWS][H_S]  hidden   (f16)
  _Float16* sW2 = sH  + ROWS * H_S;             // [Od][H_S]    W2 rows  (f16)
  float* sScale = (float*)(sW2 + Od * H_S);     // [ROWS] exp(NU*s)
  float* sB1    = sScale + ROWS;                // [Hd]
  float* sB2    = sB1 + Hd;                     // [Od]

  const int tid  = threadIdx.x;
  const int lane = tid & 31;
  const int wave = tid >> 5;
  const long long row0 = (long long)blockIdx.x * ROWS;

  // ---- weights & biases -> LDS (f32 -> f16 conversion needs VALU anyway) ----
  #pragma unroll
  for (int i = 0; i < 64; ++i) {
    const int idx = tid + i * 256;                            // 0..16383
    sW1[(idx >> 6) * XS_S + (idx & 63)]  = (_Float16)W1[idx]; // W1: [256][64]
    sW2[(idx >> 8) * H_S  + (idx & 255)] = (_Float16)W2[idx]; // W2: [64][256]
  }
  sB1[tid] = b1[tid];
  if (tid < Od) sB2[tid] = b2[tid];

  // ---- phase 1: per-row Newton solve of sum exp(-2 r_i s) x_i^2 = 1 ----
  float xv[64], tv[64];
  {
    const float4* xp = (const float4*)(x + (row0 + tid) * (long long)Nd);
    #pragma unroll
    for (int i = 0; i < 16; ++i) {
      float4 v = xp[i];
      xv[4*i+0] = v.x; xv[4*i+1] = v.y; xv[4*i+2] = v.z; xv[4*i+3] = v.w;
    }
  }
  #pragma unroll
  for (int i = 0; i < 64; ++i) tv[i] = -2.0f * r[i];   // uniform -> SGPRs

  bool zero = true;
  #pragma unroll
  for (int i = 0; i < 64; ++i) zero = zero && (fabsf(xv[i]) <= 1e-12f);

  float s = 0.0f;
  for (int it = 0; it < NEWTON_ITERS; ++it) {   // fixed trip count, per-lane freeze
    float val = -1.0f, dval = 0.0f;
    #pragma unroll
    for (int i = 0; i < 64; ++i) {
      const float w = __expf(tv[i] * s) * xv[i] * xv[i];  // exp(-2 r_i s) x_i^2
      val  += w;
      dval += tv[i] * w;
    }
    const float sd  = (dval == 0.0f) ? 1.0f : dval;
    const bool  act = (fabsf(val) >= NEWTON_TOL) && !zero;
    s -= act ? (val / sd) : 0.0f;
  }

  sScale[tid] = __expf(s);  // NU = 1
  {
    _Float16* xr = sX + tid * XS_S;
    #pragma unroll
    for (int i = 0; i < 64; ++i)
      xr[i] = (_Float16)(xv[i] * __expf(0.5f * tv[i] * s));   // x_i * exp(-r_i s)
  }
  __syncthreads();

  // ---- phase 2: GEMM1 (16x256 per wave-subtile) with fused bias+ReLU ----
  // Each wave owns rows [wave*32, wave*32+32): two 16-row WMMA subtiles.
  #pragma unroll
  for (int sub = 0; sub < 2; ++sub) {
    const int m0 = wave * 32 + sub * 16;
    const v16h a0 = ld_a_frag(sX + m0 * XS_S, XS_S, 0,  lane);
    const v16h a1 = ld_a_frag(sX + m0 * XS_S, XS_S, 32, lane);
    const int rbase = m0 + ((lane >> 4) << 3);
    for (int nt = 0; nt < 16; ++nt) {
      v8f acc = {};
      const v16h bb0 = ld_b_frag(sW1, nt * 16, XS_S, 0,  lane);
      const v16h bb1 = ld_b_frag(sW1, nt * 16, XS_S, 32, lane);
      acc = __builtin_amdgcn_wmma_f32_16x16x32_f16(false, a0, false, bb0, (short)0, acc, false, false);
      acc = __builtin_amdgcn_wmma_f32_16x16x32_f16(false, a1, false, bb1, (short)0, acc, false, false);
      const int   col  = nt * 16 + (lane & 15);
      const float bias = sB1[col];
      #pragma unroll
      for (int v = 0; v < 8; ++v) {
        const float h = acc[v] + bias;
        sH[(rbase + v) * H_S + col] = (_Float16)(h > 0.0f ? h : 0.0f);
      }
    }
  }

  // No __syncthreads needed: each wave reads back only the sH rows it wrote,
  // and a wave's DS ops retire in order (DScnt-ordered RAW handled by compiler).

  // ---- phase 3: GEMM2 (16x64 per wave-subtile) + bias + exp(s) scaling ----
  #pragma unroll
  for (int sub = 0; sub < 2; ++sub) {
    const int m0 = wave * 32 + sub * 16;
    v16h af[8];
    #pragma unroll
    for (int kt = 0; kt < 8; ++kt)
      af[kt] = ld_a_frag(sH + m0 * H_S, H_S, kt * 32, lane);
    const int rbase = m0 + ((lane >> 4) << 3);
    #pragma unroll
    for (int nt = 0; nt < 4; ++nt) {
      v8f acc = {};
      #pragma unroll
      for (int kt = 0; kt < 8; ++kt) {
        const v16h bb = ld_b_frag(sW2, nt * 16, H_S, kt * 32, lane);
        acc = __builtin_amdgcn_wmma_f32_16x16x32_f16(false, af[kt], false, bb, (short)0, acc, false, false);
      }
      const int   col  = nt * 16 + (lane & 15);
      const float bias = sB2[col];
      #pragma unroll
      for (int v = 0; v < 8; ++v) {
        const int rw = rbase + v;
        out[(row0 + rw) * (long long)Od + col] = (acc[v] + bias) * sScale[rw];
      }
    }
  }
}

extern "C" void kernel_launch(void* const* d_in, const int* in_sizes, int n_in,
                              void* d_out, int out_size, void* d_ws, size_t ws_size,
                              hipStream_t stream) {
  const float* x  = (const float*)d_in[0];
  const float* r  = (const float*)d_in[1];
  const float* W1 = (const float*)d_in[2];
  const float* b1 = (const float*)d_in[3];
  const float* W2 = (const float*)d_in[4];
  const float* b2 = (const float*)d_in[5];
  float* out = (float*)d_out;

  const int Bn = in_sizes[0] / Nd;   // 262144

  // ~239 KB dynamic LDS (within the 320 KB WGP budget); raise the cap.
  (void)hipFuncSetAttribute((const void*)aniso_mlp_fused,
                            hipFuncAttributeMaxDynamicSharedMemorySize,
                            (int)SMEM_BYTES);

  dim3 grid(Bn / ROWS), block(256);
  aniso_mlp_fused<<<grid, block, SMEM_BYTES, stream>>>(x, r, W1, b1, W2, b2, out);
}
